// PointNet2_42545946034422
// MI455X (gfx1250) — compile-verified, matrix operand order checked by reading, not run on hardware
//
#include <hip/hip_runtime.h>
#include <hip/hip_bf16.h>
#include <math.h>

typedef __attribute__((ext_vector_type(16))) _Float16 v16h;
typedef __attribute__((ext_vector_type(8)))  _Float16 v8h;
typedef __attribute__((ext_vector_type(8)))  float    v8f;

#define BB   4
#define NPTS 8192
#define SS1  2048
#define SS2  512
#define KNN  32
#define EMB  64

// ---------------------------------------------------------------------------
// Timestep embedding + 2-layer MLP (leaky_relu 0.1 between). grid=B, block=64
// ---------------------------------------------------------------------------
__global__ void tvec_kernel(const float* __restrict__ t,
                            const float* __restrict__ w1, const float* __restrict__ b1,
                            const float* __restrict__ w2, const float* __restrict__ b2,
                            float* __restrict__ tvec) {
    __shared__ float e[EMB];
    __shared__ float h1[EMB];
    const int b = blockIdx.x, i = threadIdx.x;
    const int j = i & 31;
    const float f = expf((float)j * (-logf(10000.0f) / 31.0f));
    const float a = t[b] * f;
    e[i] = (i < 32) ? sinf(a) : cosf(a);
    __syncthreads();
    float s = b1[i];
    for (int c = 0; c < EMB; ++c) s += e[c] * w1[i * EMB + c];
    h1[i] = (s >= 0.f) ? s : 0.1f * s;
    __syncthreads();
    float s2 = b2[i];
    for (int c = 0; c < EMB; ++c) s2 += h1[c] * w2[i * EMB + c];
    tvec[b * EMB + i] = s2;
}

// ---------------------------------------------------------------------------
// inputs (B,22,N) channels 0..2 -> xyz0 (B,N,3) point-major
// ---------------------------------------------------------------------------
__global__ void xyz0_kernel(const float* __restrict__ inputs, float* __restrict__ xyz0) {
    int tid = blockIdx.x * blockDim.x + threadIdx.x;
    if (tid >= BB * NPTS) return;
    int b = tid / NPTS, n = tid % NPTS;
    const float* src = inputs + (size_t)b * 22 * NPTS;
#pragma unroll
    for (int c = 0; c < 3; ++c) xyz0[(size_t)tid * 3 + c] = src[c * NPTS + n];
}

// ---------------------------------------------------------------------------
// Farthest point sampling. grid=B, block=256, LDS dist[8192] (32KB of 320KB)
// Emits current farthest BEFORE update (matches scan carry semantics).
// ---------------------------------------------------------------------------
__global__ void fps_kernel(const float* __restrict__ xyz, int* __restrict__ out,
                           int Nsrc, int S) {
    __shared__ float dist[NPTS];
    __shared__ float rv[256];
    __shared__ int   ri[256];
    __shared__ int   cur;
    const int b = blockIdx.x, t = threadIdx.x;
    const float* p = xyz + (size_t)b * Nsrc * 3;
    for (int i = t; i < Nsrc; i += 256) dist[i] = 1e10f;
    if (t == 0) cur = 0;
    __syncthreads();
    for (int s = 0; s < S; ++s) {
        const int c = cur;
        if (t == 0) out[b * S + s] = c;
        const float cx = p[c * 3 + 0], cy = p[c * 3 + 1], cz = p[c * 3 + 2];
        float bm = -1.f; int bi = 0;
        for (int i = t; i < Nsrc; i += 256) {
            float dx = p[i * 3 + 0] - cx, dy = p[i * 3 + 1] - cy, dz = p[i * 3 + 2] - cz;
            float d = dx * dx + dy * dy + dz * dz;
            float dd = fminf(dist[i], d);
            dist[i] = dd;
            if (dd > bm) { bm = dd; bi = i; }   // strided i increases -> first max kept
        }
        rv[t] = bm; ri[t] = bi;
        __syncthreads();
        for (int st = 128; st > 0; st >>= 1) {
            if (t < st) {
                if (rv[t + st] > rv[t] || (rv[t + st] == rv[t] && ri[t + st] < ri[t])) {
                    rv[t] = rv[t + st]; ri[t] = ri[t + st];
                }
            }
            __syncthreads();
        }
        if (t == 0) cur = ri[0];
        __syncthreads();
    }
}

// gather centers: out (B,S,3) = xyz[idx]
__global__ void gather_xyz_kernel(const float* __restrict__ xyz, const int* __restrict__ idx,
                                  float* __restrict__ out, int Nsrc, int S) {
    int tid = blockIdx.x * blockDim.x + threadIdx.x;
    if (tid >= BB * S) return;
    int b = tid / S;
    int j = idx[tid];
#pragma unroll
    for (int c = 0; c < 3; ++c)
        out[(size_t)tid * 3 + c] = xyz[((size_t)b * Nsrc + j) * 3 + c];
}

// ---------------------------------------------------------------------------
// Ball query: first K indices (ascending) with d2<=r2; fill with first found.
// ---------------------------------------------------------------------------
__global__ void ballquery_kernel(const float* __restrict__ xyz, const float* __restrict__ cent,
                                 int* __restrict__ out, int Nsrc, int S, float r2) {
    int tid = blockIdx.x * blockDim.x + threadIdx.x;
    if (tid >= BB * S) return;
    int b = tid / S;
    const float* c = cent + (size_t)tid * 3;
    const float cx = c[0], cy = c[1], cz = c[2];
    const float* p = xyz + (size_t)b * Nsrc * 3;
    int* o = out + (size_t)tid * KNN;
    int cnt = 0, first = 0;
    bool havefirst = false;
    for (int j = 0; j < Nsrc; ++j) {
        float dx = p[j * 3 + 0] - cx, dy = p[j * 3 + 1] - cy, dz = p[j * 3 + 2] - cz;
        float d2 = dx * dx + dy * dy + dz * dz;
        if (d2 <= r2) {
            if (!havefirst) { first = j; havefirst = true; }
            o[cnt++] = j;
            if (cnt == KNN) break;
        }
    }
    for (; cnt < KNN; ++cnt) o[cnt] = first;
}

// ---------------------------------------------------------------------------
// SA1 grouping: rows (B*S1*K) x 96 f16 = [xyz-center(3) | feats(3) | tvec(64) | pad]
// ---------------------------------------------------------------------------
__global__ void group_sa1_kernel(const float* __restrict__ inputs, const float* __restrict__ xyz0,
                                 const float* __restrict__ cent, const int* __restrict__ bq,
                                 const float* __restrict__ tvec, _Float16* __restrict__ g) {
    int tid = blockIdx.x * blockDim.x + threadIdx.x;
    const int total = BB * SS1 * KNN;
    if (tid >= total) return;
    int b = tid / (SS1 * KNN);
    int s = (tid / KNN) % SS1;
    int j = bq[tid];
    const float* c  = cent + ((size_t)b * SS1 + s) * 3;
    const float* pj = xyz0 + ((size_t)b * NPTS + j) * 3;
    const float* fin = inputs + (size_t)b * 22 * NPTS;
    const float* tv  = tvec + b * EMB;
    _Float16* row = g + (size_t)tid * 96;
#pragma unroll
    for (int q = 0; q < 3; ++q) row[q] = (_Float16)(pj[q] - c[q]);
#pragma unroll
    for (int q = 0; q < 3; ++q) row[3 + q] = (_Float16)fin[(3 + q) * NPTS + j];
    for (int q = 0; q < EMB; ++q) row[6 + q] = (_Float16)tv[q];
    for (int q = 70; q < 96; ++q) row[q] = (_Float16)0.f;
}

// ---------------------------------------------------------------------------
// SA2 grouping: rows (B*S2*K) x 160 = [xyz-center(3) | sa1_f(64) | tvec(64) | pad]
// ---------------------------------------------------------------------------
__global__ void group_sa2_kernel(const float* __restrict__ sa1f, const float* __restrict__ xyz1,
                                 const float* __restrict__ cent, const int* __restrict__ bq,
                                 const float* __restrict__ tvec, _Float16* __restrict__ g) {
    int tid = blockIdx.x * blockDim.x + threadIdx.x;
    const int total = BB * SS2 * KNN;
    if (tid >= total) return;
    int b = tid / (SS2 * KNN);
    int s = (tid / KNN) % SS2;
    int j = bq[tid];
    const float* c  = cent + ((size_t)b * SS2 + s) * 3;
    const float* pj = xyz1 + ((size_t)b * SS1 + j) * 3;
    const float* fj = sa1f + ((size_t)b * SS1 + j) * 64;
    const float* tv = tvec + b * EMB;
    _Float16* row = g + (size_t)tid * 160;
#pragma unroll
    for (int q = 0; q < 3; ++q) row[q] = (_Float16)(pj[q] - c[q]);
    for (int q = 0; q < 64; ++q) row[3 + q] = (_Float16)fj[q];
    for (int q = 0; q < EMB; ++q) row[67 + q] = (_Float16)tv[q];
    for (int q = 131; q < 160; ++q) row[q] = (_Float16)0.f;
}

// max over K=32 neighbors: y (G*K, Np) f16 -> out (G, C) f32
__global__ void maxpool_kernel(const _Float16* __restrict__ y, float* __restrict__ out,
                               int G, int Np, int C) {
    int tid = blockIdx.x * blockDim.x + threadIdx.x;
    if (tid >= G * C) return;
    int g = tid / C, c = tid % C;
    const _Float16* p = y + (size_t)g * KNN * Np + c;
    float m = -1e30f;
    for (int k = 0; k < KNN; ++k) m = fmaxf(m, (float)p[(size_t)k * Np]);
    out[tid] = m;
}

// ---------------------------------------------------------------------------
// Feature propagation: 3-NN inverse-distance interp + concat skip, f16 rows.
// h row = [interp(cf, Cc) | tvec(64) | skip(Cs) | pad to Hp]
// ---------------------------------------------------------------------------
__global__ void fp_interp_kernel(const float* __restrict__ fxyz, int Nf,
                                 const float* __restrict__ cxyz, int Nc,
                                 const float* __restrict__ cf, int Cc,
                                 const float* __restrict__ tvec,
                                 const float* __restrict__ skip, int Cs,
                                 int skipChanMajor, _Float16* __restrict__ h, int Hp) {
    int tid = blockIdx.x * blockDim.x + threadIdx.x;
    if (tid >= BB * Nf) return;
    int b = tid / Nf, n = tid % Nf;
    const float* fp = fxyz + (size_t)tid * 3;
    const float* cp = cxyz + (size_t)b * Nc * 3;
    const float fx = fp[0], fy = fp[1], fz = fp[2];
    float d0 = 1e30f, d1 = 1e30f, d2 = 1e30f;
    int   i0 = 0, i1 = 0, i2 = 0;
    for (int j = 0; j < Nc; ++j) {
        float dx = cp[j * 3 + 0] - fx, dy = cp[j * 3 + 1] - fy, dz = cp[j * 3 + 2] - fz;
        float dd = dx * dx + dy * dy + dz * dz;
        if (dd < d0)      { d2 = d1; i2 = i1; d1 = d0; i1 = i0; d0 = dd; i0 = j; }
        else if (dd < d1) { d2 = d1; i2 = i1; d1 = dd; i1 = j; }
        else if (dd < d2) { d2 = dd; i2 = j; }
    }
    float w0 = 1.f / (fmaxf(d0, 0.f) + 1e-8f);
    float w1 = 1.f / (fmaxf(d1, 0.f) + 1e-8f);
    float w2 = 1.f / (fmaxf(d2, 0.f) + 1e-8f);
    float wsum = w0 + w1 + w2;
    w0 /= wsum; w1 /= wsum; w2 /= wsum;
    const float* f0 = cf + ((size_t)b * Nc + i0) * Cc;
    const float* f1 = cf + ((size_t)b * Nc + i1) * Cc;
    const float* f2 = cf + ((size_t)b * Nc + i2) * Cc;
    _Float16* row = h + (size_t)tid * Hp;
    for (int c = 0; c < Cc; ++c)
        row[c] = (_Float16)(w0 * f0[c] + w1 * f1[c] + w2 * f2[c]);
    const float* tv = tvec + b * EMB;
    for (int q = 0; q < EMB; ++q) row[Cc + q] = (_Float16)tv[q];  // interp of constant = constant
    if (skipChanMajor) {
        const float* sk = skip + (size_t)b * Cs * Nf;
        for (int c = 0; c < Cs; ++c) row[Cc + EMB + c] = (_Float16)sk[(size_t)c * Nf + n];
    } else {
        const float* sk = skip + ((size_t)b * Nf + n) * Cs;
        for (int c = 0; c < Cs; ++c) row[Cc + EMB + c] = (_Float16)sk[c];
    }
    for (int c = Cc + EMB + Cs; c < Hp; ++c) row[c] = (_Float16)0.f;
}

// pack f32 weight (Nout,Kin) -> f16 (Np,Kp) zero-padded; bias -> padded f32
__global__ void pack_w_kernel(const float* __restrict__ W, const float* __restrict__ bsrc,
                              _Float16* __restrict__ Wp, float* __restrict__ bp,
                              int Nout, int Kin, int Np, int Kp) {
    int tid = blockIdx.x * blockDim.x + threadIdx.x;
    if (tid < Np * Kp) {
        int n = tid / Kp, k = tid % Kp;
        Wp[tid] = (n < Nout && k < Kin) ? (_Float16)W[(size_t)n * Kin + k] : (_Float16)0.f;
    }
    if (tid < Np) bp[tid] = (tid < Nout) ? bsrc[tid] : 0.f;
}

// ---------------------------------------------------------------------------
// WMMA GEMM: Y[M,Np] = relu(X[M,Kp] * Wp[Np,Kp]^T + b). Each wave owns one
// 16-row A strip and NT consecutive 16-col N-tiles: A fragment is loaded once
// per K-step and fed to NT back-to-back v_wmma_f32_16x16x32_f16, raising
// WMMA-per-byte 4x vs one-tile-per-wave. 16-bit operand striping per ISA
// 7.12.2 (lanes 0-15: K {0..7,16..23}; lanes 16-31: K {8..15,24..31}).
// M, Kp(%32), Np(%(16*NT)) padded -> EXEC all-ones at every WMMA; the only
// guard is a wave-uniform early return.
// ---------------------------------------------------------------------------
template <int NT>
__global__ void wmma_gemm_kernel(const _Float16* __restrict__ X,
                                 const _Float16* __restrict__ Wp,
                                 const float* __restrict__ bp,
                                 _Float16* __restrict__ Yh,
                                 float* __restrict__ Yf,
                                 int M, int Kp, int Np, int relu, int outf32) {
    const int lane = threadIdx.x & 31;
    const long wave = ((long)blockIdx.x * blockDim.x + threadIdx.x) >> 5;
    const int groupsN = (Np >> 4) / NT;
    const long totalWaves = (long)(M >> 4) * groupsN;
    if (wave >= totalWaves) return;          // wave-uniform exit
    const int tm  = (int)(wave / groupsN);
    const int tn0 = (int)(wave % groupsN) * NT;
    const int lr = lane & 15;
    const int kb = (lane >> 4) << 3;         // 0 or 8
    const _Float16* __restrict__ xrow = X + (size_t)(tm * 16 + lr) * Kp;
    const _Float16* wrow[NT];
#pragma unroll
    for (int i = 0; i < NT; ++i)
        wrow[i] = Wp + (size_t)((tn0 + i) * 16 + lr) * Kp;
    v8f acc[NT];
#pragma unroll
    for (int i = 0; i < NT; ++i) acc[i] = (v8f){};
    for (int k0 = 0; k0 < Kp; k0 += 32) {
        __builtin_prefetch(xrow + k0 + 64, 0, 1);   // global_prefetch_b8
        union { v16h v; v8h h[2]; } a;
        a.h[0] = *(const v8h*)(xrow + k0 + kb);
        a.h[1] = *(const v8h*)(xrow + k0 + 16 + kb);
#pragma unroll
        for (int i = 0; i < NT; ++i) {
            union { v16h v; v8h h[2]; } b;
            b.h[0] = *(const v8h*)(wrow[i] + k0 + kb);
            b.h[1] = *(const v8h*)(wrow[i] + k0 + 16 + kb);
            acc[i] = __builtin_amdgcn_wmma_f32_16x16x32_f16(false, a.v, false, b.v,
                                                            (short)0, acc[i], false, false);
        }
    }
    const int mbase = tm * 16 + ((lane >> 4) << 3);
#pragma unroll
    for (int i = 0; i < NT; ++i) {
        const int col = (tn0 + i) * 16 + lr;
        const float bias = bp[col];
#pragma unroll
        for (int r = 0; r < 8; ++r) {
            float v = acc[i][r] + bias;
            if (relu) v = v > 0.f ? v : 0.f;
            size_t o = (size_t)(mbase + r) * Np + col;
            if (outf32) Yf[o] = v; else Yh[o] = (_Float16)v;
        }
    }
}

// ycls (B*N,64) f32 -> out (B,50,N) f32
__global__ void out_kernel(const float* __restrict__ ycls, float* __restrict__ out) {
    int tid = blockIdx.x * blockDim.x + threadIdx.x;
    const int total = BB * 50 * NPTS;
    if (tid >= total) return;
    int b = tid / (50 * NPTS);
    int r = tid % (50 * NPTS);
    int o = r / NPTS, n = r % NPTS;
    out[tid] = ycls[((size_t)b * NPTS + n) * 64 + o];
}

// ---------------------------------------------------------------------------
extern "C" void kernel_launch(void* const* d_in, const int* in_sizes, int n_in,
                              void* d_out, int out_size, void* d_ws, size_t ws_size,
                              hipStream_t stream) {
    const float* inputs = (const float*)d_in[0];
    const float* t      = (const float*)d_in[1];
    const float* P[24];
    for (int i = 0; i < 24; ++i) P[i] = (const float*)d_in[2 + i];
    // P: 0..3 emb(w1,b1,w2,b2) 4..7 sa1 8..11 sa2 12..15 fp1 16..19 fp2 20..23 cls

    char* ws = (char*)d_ws;
    size_t off = 0;
    auto alloc = [&](size_t bytes) -> char* {
        char* p = ws + off;
        off = (off + bytes + 255) & ~(size_t)255;
        return p;
    };
    float* tvec   = (float*)alloc(BB * EMB * 4);
    float* xyz0   = (float*)alloc((size_t)BB * NPTS * 3 * 4);
    int*   fps1   = (int*)  alloc((size_t)BB * SS1 * 4);
    float* xyz1   = (float*)alloc((size_t)BB * SS1 * 3 * 4);
    int*   bq1    = (int*)  alloc((size_t)BB * SS1 * KNN * 4);
    int*   fps2   = (int*)  alloc((size_t)BB * SS2 * 4);
    float* xyz2   = (float*)alloc((size_t)BB * SS2 * 3 * 4);
    int*   bq2    = (int*)  alloc((size_t)BB * SS2 * KNN * 4);
    float* sa1f   = (float*)alloc((size_t)BB * SS1 * 64 * 4);
    float* sa2f   = (float*)alloc((size_t)BB * SS2 * 128 * 4);
    float* fp1f   = (float*)alloc((size_t)BB * SS1 * 128 * 4);
    // packed weights
    struct WP { _Float16* w; float* b; int N, K, Np, Kp; };
    int wn[10]  = {32, 64, 64, 128, 128, 128, 128, 64, 128, 50};
    int wk[10]  = {70, 32, 131, 64, 256, 128, 214, 128, 64, 128};
    int wnp[10] = {32, 64, 64, 128, 128, 128, 128, 64, 128, 64};
    int wkp[10] = {96, 32, 160, 64, 256, 128, 224, 128, 64, 128};
    const float* wsrc[10] = {P[4], P[6], P[8], P[10], P[12], P[14], P[16], P[18], P[20], P[22]};
    const float* bsrc[10] = {P[5], P[7], P[9], P[11], P[13], P[15], P[17], P[19], P[21], P[23]};
    WP wp[10];
    for (int i = 0; i < 10; ++i) {
        wp[i].N = wn[i]; wp[i].K = wk[i]; wp[i].Np = wnp[i]; wp[i].Kp = wkp[i];
        wp[i].w = (_Float16*)alloc((size_t)wnp[i] * wkp[i] * 2);
        wp[i].b = (float*)alloc((size_t)wnp[i] * 4);
    }
    char* slab0 = alloc((size_t)50 << 20);
    char* slab1 = alloc((size_t)33 << 20);
    char* slab2 = alloc((size_t)33 << 20);
    (void)ws_size; (void)n_in; (void)in_sizes; (void)out_size;

    auto grid = [](long total, int block) { return (unsigned)((total + block - 1) / block); };

    // --- weights + t embedding + xyz transpose ---
    for (int i = 0; i < 10; ++i) {
        long tot = (long)wp[i].Np * wp[i].Kp;
        pack_w_kernel<<<grid(tot, 256), 256, 0, stream>>>(wsrc[i], bsrc[i], wp[i].w, wp[i].b,
                                                          wp[i].N, wp[i].K, wp[i].Np, wp[i].Kp);
    }
    tvec_kernel<<<BB, EMB, 0, stream>>>(t, P[0], P[1], P[2], P[3], tvec);
    xyz0_kernel<<<grid((long)BB * NPTS, 256), 256, 0, stream>>>(inputs, xyz0);

    auto gemm = [&](const _Float16* X, int wi, _Float16* Yh, float* Yf, int M, int relu) {
        int tilesN = wp[wi].Np / 16;
        if ((tilesN & 3) == 0) {
            long waves = (long)(M / 16) * (tilesN / 4);
            wmma_gemm_kernel<4><<<grid(waves * 32, 256), 256, 0, stream>>>(
                X, wp[wi].w, wp[wi].b, Yh, Yf, M, wp[wi].Kp, wp[wi].Np, relu, Yf != nullptr);
        } else {
            long waves = (long)(M / 16) * (tilesN / 2);
            wmma_gemm_kernel<2><<<grid(waves * 32, 256), 256, 0, stream>>>(
                X, wp[wi].w, wp[wi].b, Yh, Yf, M, wp[wi].Kp, wp[wi].Np, relu, Yf != nullptr);
        }
    };

    // --- SA1 ---
    fps_kernel<<<BB, 256, 0, stream>>>(xyz0, fps1, NPTS, SS1);
    gather_xyz_kernel<<<grid((long)BB * SS1, 256), 256, 0, stream>>>(xyz0, fps1, xyz1, NPTS, SS1);
    ballquery_kernel<<<grid((long)BB * SS1, 256), 256, 0, stream>>>(xyz0, xyz1, bq1, NPTS, SS1, 0.01f);
    _Float16* g1 = (_Float16*)slab0;                         // 262144 x 96
    group_sa1_kernel<<<grid((long)BB * SS1 * KNN, 256), 256, 0, stream>>>(inputs, xyz0, xyz1, bq1, tvec, g1);
    _Float16* y1a = (_Float16*)slab1;                        // 262144 x 32
    gemm(g1, 0, y1a, nullptr, BB * SS1 * KNN, 1);
    _Float16* y1b = (_Float16*)slab2;                        // 262144 x 64
    gemm(y1a, 1, y1b, nullptr, BB * SS1 * KNN, 1);
    maxpool_kernel<<<grid((long)BB * SS1 * 64, 256), 256, 0, stream>>>(y1b, sa1f, BB * SS1, 64, 64);

    // --- SA2 ---
    fps_kernel<<<BB, 256, 0, stream>>>(xyz1, fps2, SS1, SS2);
    gather_xyz_kernel<<<grid((long)BB * SS2, 256), 256, 0, stream>>>(xyz1, fps2, xyz2, SS1, SS2);
    ballquery_kernel<<<grid((long)BB * SS2, 256), 256, 0, stream>>>(xyz1, xyz2, bq2, SS1, SS2, 0.04f);
    _Float16* g2 = (_Float16*)slab0;                         // 65536 x 160
    group_sa2_kernel<<<grid((long)BB * SS2 * KNN, 256), 256, 0, stream>>>(sa1f, xyz1, xyz2, bq2, tvec, g2);
    _Float16* y2a = (_Float16*)slab1;                        // 65536 x 64
    gemm(g2, 2, y2a, nullptr, BB * SS2 * KNN, 1);
    _Float16* y2b = (_Float16*)slab2;                        // 65536 x 128
    gemm(y2a, 3, y2b, nullptr, BB * SS2 * KNN, 1);
    maxpool_kernel<<<grid((long)BB * SS2 * 128, 256), 256, 0, stream>>>(y2b, sa2f, BB * SS2, 128, 128);

    // --- FP1: 512 -> 2048, h = [interp(sa2f 128 + tvec 64) | skip sa1f 64] ---
    _Float16* h1 = (_Float16*)slab0;                         // 8192 x 256
    fp_interp_kernel<<<grid((long)BB * SS1, 256), 256, 0, stream>>>(
        xyz1, SS1, xyz2, SS2, sa2f, 128, tvec, sa1f, 64, 0, h1, 256);
    _Float16* yf1a = (_Float16*)slab1;                       // 8192 x 128
    gemm(h1, 4, yf1a, nullptr, BB * SS1, 1);
    gemm(yf1a, 5, nullptr, fp1f, BB * SS1, 1);               // fp1f (B*2048,128) f32

    // --- FP2: 2048 -> 8192, h = [interp(fp1f 128 + tvec 64) | skip inputs 22] ---
    _Float16* h2 = (_Float16*)slab0;                         // 32768 x 224
    fp_interp_kernel<<<grid((long)BB * NPTS, 256), 256, 0, stream>>>(
        xyz0, NPTS, xyz1, SS1, fp1f, 128, tvec, inputs, 22, 1, h2, 224);
    _Float16* yf2a = (_Float16*)slab1;                       // 32768 x 128
    gemm(h2, 6, yf2a, nullptr, BB * NPTS, 1);
    _Float16* curf = (_Float16*)slab2;                       // 32768 x 64
    gemm(yf2a, 7, curf, nullptr, BB * NPTS, 1);

    // --- classifier ---
    _Float16* hcls = (_Float16*)slab0;                       // 32768 x 128
    gemm(curf, 8, hcls, nullptr, BB * NPTS, 1);
    float* ycls = (float*)slab1;                             // 32768 x 64 f32
    gemm(hcls, 9, nullptr, ycls, BB * NPTS, 0);
    out_kernel<<<grid((long)BB * 50 * NPTS, 256), 256, 0, stream>>>(ycls, (float*)d_out);
}
